// MultiheadAttentionRPR_9423158247861
// MI455X (gfx1250) — matrix operand order repaired and checked by reading, not compiled, not run on hardware
//
#include <hip/hip_runtime.h>
#include <hip/hip_bf16.h>

// ---------------------------------------------------------------------------
// MultiheadAttention with relative position representations (Music-Transformer
// skew), fused flash-attention style for MI455X (gfx1250, wave32, WMMA).
//   L=2048, B=2, E=512, H=8, D=64, BH=B*H=16
// All matmuls use v_wmma_f32_16x16x32_bf16 (f32 accumulate).
// ---------------------------------------------------------------------------

#define LQ   2048
#define BB   2
#define EE   512
#define HH   8
#define DD   64
#define BH   16
#define MR   4096   // L*B rows of the token-major GEMMs
#define K3E  1536   // 3*E

typedef __attribute__((ext_vector_type(16))) __bf16 v16bf;
typedef __attribute__((ext_vector_type(8)))  float  v8f;

// round-to-nearest-even f32 -> bf16 (as raw u16)
__device__ __forceinline__ unsigned short f2bf(float f) {
    union { float f; unsigned int u; } v; v.f = f;
    unsigned int u = v.u;
    return (unsigned short)((u + 0x7FFFu + ((u >> 16) & 1u)) >> 16);
}

__device__ __forceinline__ v8f wmma_bf16(v16bf a, v16bf b, v8f c) {
    // (neg_a, A, neg_b, B, c_mod, C, reuse_a, reuse_b)
    return __builtin_amdgcn_wmma_f32_16x16x32_bf16(false, a, false, b,
                                                   (short)0, c, false, false);
}

// A operand (16x32, M x K) from row-major bf16 [rows][ld]:
//   lanes 0-15 : row = lane,     halfs 0..7 = K k0..k0+7,  8..15 = k0+16..k0+23
//   lanes 16-31: row = lane-16,  halfs 0..7 = K k0+8..+15, 8..15 = k0+24..k0+31
__device__ __forceinline__ v16bf load_A(const unsigned short* base, int ld,
                                        int row0, int k0) {
    int lane = threadIdx.x & 31;
    const unsigned short* p = base + (size_t)(row0 + (lane & 15)) * ld
                                   + k0 + ((lane >> 4) ? 8 : 0);
    union { uint4 q[2]; v16bf v; } u;
    u.q[0] = *(const uint4*)(p);
    u.q[1] = *(const uint4*)(p + 16);
    return u.v;
}

// B operand (32x16, K x N) where memory holds B^T row-major: src[n][k].
//   lanes 0-15 : col n = lane,    halfs 0..7 = K k0..k0+7,  8..15 = k0+16..k0+23
//   lanes 16-31: col n = lane-16, halfs 0..7 = K k0+8..+15, 8..15 = k0+24..k0+31
__device__ __forceinline__ v16bf load_B(const unsigned short* base, int ld,
                                        int n0, int k0) {
    int lane = threadIdx.x & 31;
    const unsigned short* p = base + (size_t)(n0 + (lane & 15)) * ld
                                   + k0 + ((lane >> 4) ? 8 : 0);
    union { uint4 q[2]; v16bf v; } u;
    u.q[0] = *(const uint4*)(p);
    u.q[1] = *(const uint4*)(p + 16);
    return u.v;
}

// B operand from Er (rows clamped to L-1; clamped elements are masked later)
__device__ __forceinline__ v16bf load_B_er(const unsigned short* er,
                                           int n0, int k0) {
    int lane = threadIdx.x & 31;
    int n = n0 + (lane & 15);
    if (n > LQ - 1) n = LQ - 1;
    const unsigned short* p = er + (size_t)n * DD + k0 + ((lane >> 4) ? 8 : 0);
    union { uint4 q[2]; v16bf v; } u;
    u.q[0] = *(const uint4*)(p);
    u.q[1] = *(const uint4*)(p + 16);
    return u.v;
}

// ---------------------------------------------------------------------------
__global__ __launch_bounds__(256) void convert_f32_bf16(const float* __restrict__ in,
                                                        unsigned short* __restrict__ out,
                                                        int n) {
    int i = blockIdx.x * 256 + threadIdx.x;
    if (i < n) out[i] = f2bf(in[i]);
}

// ---------------------------------------------------------------------------
// qkv = x @ W_in^T + b_in ; scatter q(*0.125)/k head-major, v transposed (DxL)
// grid (MR/32, K3E/64), 256 threads = 8 waves (2 x 4 tiles of 16x16)
// ---------------------------------------------------------------------------
__global__ __launch_bounds__(256) void qkv_gemm(
        const unsigned short* __restrict__ xbf,
        const unsigned short* __restrict__ wbf,
        const float* __restrict__ bias,
        unsigned short* __restrict__ qbf,
        unsigned short* __restrict__ kbf,
        unsigned short* __restrict__ vT) {
    int wave = threadIdx.x >> 5;
    int lane = threadIdx.x & 31;
    int m0 = blockIdx.x * 32 + (wave >> 2) * 16;
    int n0 = blockIdx.y * 64 + (wave & 3) * 16;

    v8f acc = {};
#pragma unroll
    for (int k0 = 0; k0 < EE; k0 += 32)
        acc = wmma_bf16(load_A(xbf, EE, m0, k0), load_B(wbf, EE, n0, k0), acc);

    int col = n0 + (lane & 15);
    float bval = bias[col];
    int sec = col >> 9;        // 0:q 1:k 2:v
    int nn  = col & 511;
    int h = nn >> 6, d = nn & 63;
#pragma unroll
    for (int v = 0; v < 8; ++v) {
        int row = m0 + v + ((lane >> 4) << 3);
        int l = row >> 1, b_ = row & 1;          // row = l*B + b
        int bh = b_ * HH + h;
        float val = acc[v] + bval;
        if (sec == 0)      qbf[((size_t)bh * LQ + l) * DD + d] = f2bf(val * 0.125f);
        else if (sec == 1) kbf[((size_t)bh * LQ + l) * DD + d] = f2bf(val);
        else               vT [((size_t)bh * DD + d) * LQ + l] = f2bf(val);
    }
}

// ---------------------------------------------------------------------------
// Fused attention with relative-position skew. One wave per (16-row tile, bh).
// grid (L/16, BH), 32 threads.
// ---------------------------------------------------------------------------
__global__ __launch_bounds__(32) void attn_kernel(
        const unsigned short* __restrict__ qbf,
        const unsigned short* __restrict__ kbf,
        const unsigned short* __restrict__ vT,
        const unsigned short* __restrict__ erbf,
        unsigned short* __restrict__ attnbuf) {
    __shared__ __align__(16) float          ldsE[16][48];
    __shared__ __align__(16) unsigned short ldsP[16][32];

    int lane = threadIdx.x & 31;
    int half = lane >> 4;
    int cn   = lane & 15;
    int l0 = blockIdx.x * 16;
    int bh = blockIdx.y;
    int b_ = bh >> 3;            // bh = b*H + h
    int h  = bh & 7;

    const unsigned short* qh = qbf + (size_t)bh * LQ * DD;
    const unsigned short* kh = kbf + (size_t)bh * LQ * DD;
    const unsigned short* vh = vT  + (size_t)bh * DD * LQ;

    // Q tile (16x64) kept in registers: two K-halves
    v16bf qa0 = load_A(qh, DD, l0, 0);
    v16bf qa1 = load_A(qh, DD, l0, 32);

    v8f o0 = {}, o1 = {}, o2 = {}, o3 = {};
    float mrun[8], lrun[8];
#pragma unroll
    for (int v = 0; v < 8; ++v) { mrun[v] = -1e30f; lrun[v] = 0.f; }

    for (int m0 = 0; m0 < LQ; m0 += 32) {
        // ---- S = Q @ K^T (16x32) ----
        v8f s0 = {}, s1 = {};
        s0 = wmma_bf16(qa0, load_B(kh, DD, m0,      0),  s0);
        s0 = wmma_bf16(qa1, load_B(kh, DD, m0,      32), s0);
        s1 = wmma_bf16(qa0, load_B(kh, DD, m0 + 16, 0),  s1);
        s1 = wmma_bf16(qa1, load_B(kh, DD, m0 + 16, 32), s1);

        // ---- relative bias band: skew(QE)[l,m] = q[l]·Er[L-1-(l-m)], m<=l ----
        if (m0 <= l0 + 15) {
            int j0 = (LQ - 16 - l0) + m0;              // >= 0 always
            v8f e0 = {}, e1 = {}, e2 = {};
            e0 = wmma_bf16(qa0, load_B_er(erbf, j0,      0),  e0);
            e0 = wmma_bf16(qa1, load_B_er(erbf, j0,      32), e0);
            e1 = wmma_bf16(qa0, load_B_er(erbf, j0 + 16, 0),  e1);
            e1 = wmma_bf16(qa1, load_B_er(erbf, j0 + 16, 32), e1);
            e2 = wmma_bf16(qa0, load_B_er(erbf, j0 + 32, 0),  e2);
            e2 = wmma_bf16(qa1, load_B_er(erbf, j0 + 32, 32), e2);
#pragma unroll
            for (int v = 0; v < 8; ++v) {
                int r = v + half * 8;
                ldsE[r][cn]      = e0[v];
                ldsE[r][16 + cn] = e1[v];
                ldsE[r][32 + cn] = e2[v];
            }
            __syncthreads();           // single-wave WG: compiler/mem fence
#pragma unroll
            for (int v = 0; v < 8; ++v) {
                int r = v + half * 8;
                int lg = l0 + r;
                if (m0 + cn      <= lg) s0[v] += ldsE[r][15 - r + cn];
                if (m0 + 16 + cn <= lg) s1[v] += ldsE[r][31 - r + cn];
            }
            __syncthreads();
        }

        // ---- online softmax ----
#pragma unroll
        for (int v = 0; v < 8; ++v) {
            float mx = fmaxf(s0[v], s1[v]);
#pragma unroll
            for (int off = 8; off >= 1; off >>= 1)
                mx = fmaxf(mx, __shfl_xor(mx, off, 16));
            float mnew = fmaxf(mrun[v], mx);
            float sc = __expf(mrun[v] - mnew);
            mrun[v] = mnew;
            s0[v] = __expf(s0[v] - mnew);
            s1[v] = __expf(s1[v] - mnew);
            float rs = s0[v] + s1[v];
#pragma unroll
            for (int off = 8; off >= 1; off >>= 1)
                rs += __shfl_xor(rs, off, 16);
            lrun[v] = lrun[v] * sc + rs;
            o0[v] *= sc; o1[v] *= sc; o2[v] *= sc; o3[v] *= sc;
        }

        // ---- P (C-layout) -> LDS bf16 -> A-layout ----
#pragma unroll
        for (int v = 0; v < 8; ++v) {
            int r = v + half * 8;
            ldsP[r][cn]      = f2bf(s0[v]);
            ldsP[r][16 + cn] = f2bf(s1[v]);
        }
        __syncthreads();
        v16bf pa;
        {
            int r  = lane & 15;
            int kk = half ? 8 : 0;
            union { uint4 q[2]; v16bf v; } u;
            u.q[0] = *(const uint4*)&ldsP[r][kk];
            u.q[1] = *(const uint4*)&ldsP[r][kk + 16];
            pa = u.v;
        }
        __syncthreads();

        // ---- O += P @ V   (V accessed via D-major transpose) ----
        o0 = wmma_bf16(pa, load_B(vh, LQ, 0,  m0), o0);
        o1 = wmma_bf16(pa, load_B(vh, LQ, 16, m0), o1);
        o2 = wmma_bf16(pa, load_B(vh, LQ, 32, m0), o2);
        o3 = wmma_bf16(pa, load_B(vh, LQ, 48, m0), o3);
    }

    // normalize + write bf16 (row-major (L*B, E), m = l*B + b, col = h*64 + d)
#pragma unroll
    for (int v = 0; v < 8; ++v) {
        float inv = 1.0f / lrun[v];
        int r = v + half * 8;
        int l = l0 + r;
        unsigned short* dst = attnbuf + ((size_t)l * BB + b_) * EE + h * DD;
        dst[cn]      = f2bf(o0[v] * inv);
        dst[16 + cn] = f2bf(o1[v] * inv);
        dst[32 + cn] = f2bf(o2[v] * inv);
        dst[48 + cn] = f2bf(o3[v] * inv);
    }
}

// ---------------------------------------------------------------------------
// out = attn @ W_out^T + b_out (f32 output). grid (MR/32, E/64), 256 threads.
// ---------------------------------------------------------------------------
__global__ __launch_bounds__(256) void out_gemm(
        const unsigned short* __restrict__ abuf,
        const unsigned short* __restrict__ wbf,
        const float* __restrict__ bias,
        float* __restrict__ out) {
    int wave = threadIdx.x >> 5;
    int lane = threadIdx.x & 31;
    int m0 = blockIdx.x * 32 + (wave >> 2) * 16;
    int n0 = blockIdx.y * 64 + (wave & 3) * 16;

    v8f acc = {};
#pragma unroll
    for (int k0 = 0; k0 < EE; k0 += 32)
        acc = wmma_bf16(load_A(abuf, EE, m0, k0), load_B(wbf, EE, n0, k0), acc);

    int col = n0 + (lane & 15);
    float bval = bias[col];
#pragma unroll
    for (int v = 0; v < 8; ++v) {
        int row = m0 + v + ((lane >> 4) << 3);
        out[(size_t)row * EE + col] = acc[v] + bval;
    }
}

// ---------------------------------------------------------------------------
extern "C" void kernel_launch(void* const* d_in, const int* in_sizes, int n_in,
                              void* d_out, int out_size, void* d_ws, size_t ws_size,
                              hipStream_t stream) {
    (void)in_sizes; (void)n_in; (void)out_size; (void)ws_size;
    const float* x     = (const float*)d_in[0];
    const float* w_in  = (const float*)d_in[1];
    const float* b_in  = (const float*)d_in[2];
    const float* w_out = (const float*)d_in[3];
    const float* b_out = (const float*)d_in[4];
    const float* er    = (const float*)d_in[5];

    char* ws = (char*)d_ws;
    size_t off = 0;
    unsigned short* xbf   = (unsigned short*)(ws + off); off += (size_t)MR  * EE * 2;
    unsigned short* wibf  = (unsigned short*)(ws + off); off += (size_t)K3E * EE * 2;
    unsigned short* wobf  = (unsigned short*)(ws + off); off += (size_t)EE  * EE * 2;
    unsigned short* erbf  = (unsigned short*)(ws + off); off += (size_t)LQ  * DD * 2;
    unsigned short* qbf   = (unsigned short*)(ws + off); off += (size_t)BH * LQ * DD * 2;
    unsigned short* kbf   = (unsigned short*)(ws + off); off += (size_t)BH * LQ * DD * 2;
    unsigned short* vT    = (unsigned short*)(ws + off); off += (size_t)BH * DD * LQ * 2;
    unsigned short* abuf  = (unsigned short*)(ws + off); off += (size_t)MR * EE * 2;

    convert_f32_bf16<<<(MR * EE  + 255) / 256, 256, 0, stream>>>(x,     xbf,  MR * EE);
    convert_f32_bf16<<<(K3E * EE + 255) / 256, 256, 0, stream>>>(w_in,  wibf, K3E * EE);
    convert_f32_bf16<<<(EE * EE  + 255) / 256, 256, 0, stream>>>(w_out, wobf, EE * EE);
    convert_f32_bf16<<<(LQ * DD  + 255) / 256, 256, 0, stream>>>(er,    erbf, LQ * DD);

    qkv_gemm   <<<dim3(MR / 32, K3E / 64), 256, 0, stream>>>(xbf, wibf, b_in, qbf, kbf, vT);
    attn_kernel<<<dim3(LQ / 16, BH),        32, 0, stream>>>(qbf, kbf, vT, erbf, abuf);
    out_gemm   <<<dim3(MR / 32, EE / 64),  256, 0, stream>>>(abuf, wobf, b_out, (float*)d_out);
}
